// MultiHeadAttention_7765300871641
// MI455X (gfx1250) — matrix operand order repaired and checked by reading, not compiled, not run on hardware
//
#include <hip/hip_runtime.h>

// Problem constants (match reference)
constexpr int CB  = 2;     // batch
constexpr int CS  = 2048;  // sequence
constexpr int CHID = 1024; // hidden
constexpr int CH  = 16;    // heads
constexpr int CD  = 64;    // head dim

typedef __attribute__((ext_vector_type(16))) __bf16 v16bf;
typedef __attribute__((ext_vector_type(8)))  __bf16 v8bf;
typedef __attribute__((ext_vector_type(8)))  float  v8f;
typedef __attribute__((ext_vector_type(4)))  unsigned int v4u;
typedef __attribute__((ext_vector_type(8)))  int v8i;
typedef __attribute__((ext_vector_type(4)))  int v4i;

__device__ __forceinline__ v16bf cat8(v8bf a, v8bf b) {
  return __builtin_shufflevector(a, b, 0,1,2,3,4,5,6,7,8,9,10,11,12,13,14,15);
}

__device__ __forceinline__ v8f wmma_bf16(v16bf a, v16bf b, v8f c) {
  // D = A(16x32 bf16) * B(32x16 bf16) + C(16x16 f32)
  return __builtin_amdgcn_wmma_f32_16x16x32_bf16(
      /*neg_a=*/false, a, /*neg_b=*/false, b,
      /*c_mod=*/(short)0, c, /*reuse_a=*/false, /*reuse_b=*/false);
}

// ---------------------------------------------------------------------------
// TDM: issue a 2D tile load (elem = 2 bytes) global -> LDS.
//   tile: tile_n rows x tile_k elems, row stride = stride0 elems,
//   tensor bounds tdim0 x tdim1 (elems), packed contiguously into LDS.
// D# layout per CDNA5 ISA 8.3/8.4; groups 2/3 zero (tile_dim2/3 = 0 = unused).
// clang-23 lane: 6-arg builtin (g0 x4u, g1 x8i, g2 x4i, g3 x4i, x8i, cpol).
// ---------------------------------------------------------------------------
__device__ __forceinline__ void tdm_load_2d(unsigned lds_off, const __bf16* gptr,
                                            unsigned tile_k, unsigned tile_n,
                                            unsigned tdim0, unsigned tdim1,
                                            unsigned stride0) {
  unsigned long long ga = (unsigned long long)(uintptr_t)gptr;
  v4u g0;
  g0[0] = 1u;                                         // count=1 (valid user D#)
  g0[1] = lds_off;                                    // lds_addr (bytes)
  g0[2] = (unsigned)(ga & 0xffffffffu);               // global_addr[31:0]
  g0[3] = (unsigned)((ga >> 32) & 0x01ffffffu)        // global_addr[56:32]
        | (2u << 30);                                 // type = 2 ("image")
  v8i g1;
  g1[0] = (int)(1u << 16);                            // data_size=1 -> 2 bytes
  g1[1] = (int)((tdim0 & 0xffffu) << 16);             // tensor_dim0[15:0]
  g1[2] = (int)(((tdim0 >> 16) & 0xffffu) | ((tdim1 & 0xffffu) << 16));
  g1[3] = (int)(((tdim1 >> 16) & 0xffffu) | ((tile_k & 0xffffu) << 16)); // tile_dim0
  g1[4] = (int)(tile_n & 0xffffu);                    // tile_dim1 (tile_dim2=0)
  g1[5] = (int)stride0;                               // tensor_dim0_stride[31:0]
  g1[6] = 0;                                          // stride hi, dim1_stride lo
  g1[7] = 0;
  v4i z4 = {0, 0, 0, 0};
  v8i z8 = {0, 0, 0, 0, 0, 0, 0, 0};
  __builtin_amdgcn_tensor_load_to_lds(g0, g1, z4, z4, z8, 0);
}

// ---------------------------------------------------------------------------
// f32 -> bf16 conversion
// ---------------------------------------------------------------------------
__global__ void cvt_f32_bf16(const float* __restrict__ src, __bf16* __restrict__ dst, int n) {
  int i = blockIdx.x * blockDim.x + threadIdx.x;
  int stride = gridDim.x * blockDim.x;
  for (; i < n; i += stride) dst[i] = (__bf16)src[i];
}

// ---------------------------------------------------------------------------
// Projection GEMM:  C[M=B*S, N=HID] = A_bf16 * W_bf16^T   (W row-major == Bt)
// W tile (64x32 per K-step) staged block-wide into LDS by the Tensor Data
// Mover (double-buffered, wave0 issues, TENSORcnt + barrier publish).
// mode 0: RoPE, store bf16 [B][H][S][D]
// mode 1: no RoPE, store bf16 transposed [B][H][D][S]  (for V)
// mode 2: store f32 row-major [M][HID] (final output projection)
// Block: 256 threads = 8 waves; wave computes 16(M) x 64(N); grid (M/128, HID/64)
// ---------------------------------------------------------------------------
__global__ void __launch_bounds__(256) proj_gemm(
    const __bf16* __restrict__ A, const __bf16* __restrict__ W,
    __bf16* __restrict__ obf, float* __restrict__ ofl,
    const float* __restrict__ fcos, const float* __restrict__ fsin, int mode) {
  __shared__ __align__(32) __bf16 wtile[2][64 * 32];  // 2 x 4KB double buffer

  const int lane = threadIdx.x & 31;
  const int wv   = threadIdx.x >> 5;
  const int lo   = lane & 15;
  const int hi   = lane >> 4;
  const int row0 = blockIdx.x * 128 + wv * 16;
  const int n0   = blockIdx.y * 64;

  v8f acc[4] = {};

  const __bf16* arow = A + (size_t)(row0 + lo) * CHID + hi * 8;
  const __bf16* wbase = W + (size_t)n0 * CHID;

  const unsigned ldsbuf[2] = {
      (unsigned)(uintptr_t)(&wtile[0][0]),
      (unsigned)(uintptr_t)(&wtile[1][0])};

  if (wv == 0) {  // prime the pipeline: tile for kk=0
    tdm_load_2d(ldsbuf[0], wbase, /*tile_k=*/32, /*tile_n=*/64,
                /*tdim0=*/CHID, /*tdim1=*/CHID, /*stride0=*/CHID);
  }

  int cur = 0;
  for (int kk = 0; kk < CHID; kk += 32, cur ^= 1) {
    if (wv == 0) __builtin_amdgcn_s_wait_tensorcnt((short)0);  // tile kk landed
    __syncthreads();  // publish tile kk; also: buffer cur^1 free (compute kk-32 done)
    if (wv == 0 && kk + 32 < CHID) {
      tdm_load_2d(ldsbuf[cur ^ 1], wbase + kk + 32, 32, 64, CHID, CHID, CHID);
    }
    __builtin_prefetch(arow + kk + 256, 0, 0);
    v16bf a = cat8(*(const v8bf*)(arow + kk), *(const v8bf*)(arow + kk + 16));
    const __bf16* wl = &wtile[cur][0];
#pragma unroll
    for (int t = 0; t < 4; ++t) {
      v16bf b = cat8(*(const v8bf*)(wl + (t * 16 + lo) * 32 + hi * 8), *(const v8bf*)(wl + (t * 16 + lo) * 32 + hi * 8 + 16));
      acc[t] = wmma_bf16(a, b, acc[t]);
    }
  }

  if (mode == 0) {  // RoPE then [B][H][S][D]
#pragma unroll
    for (int t = 0; t < 4; ++t) {
      const int n = n0 + t * 16 + lo;
      const int h = n >> 6, d = n & 63, fi = d >> 1;
#pragma unroll
      for (int r = 0; r < 8; ++r) {
        const int M = row0 + r + hi * 8;
        const int srow = M & (CS - 1);
        const int bpos = M >> 11;
        float x = acc[t][r];
        float y = __shfl_xor(x, 1, 32);
        float cs = fcos[srow * (CD / 2) + fi];
        float sn = fsin[srow * (CD / 2) + fi];
        float outv = (d & 1) ? (y * sn + x * cs) : (x * cs - y * sn);
        obf[(((size_t)(bpos * CH + h) * CS + srow) * CD + d)] = (__bf16)outv;
      }
    }
  } else if (mode == 1) {  // V transposed [B][H][D][S]
#pragma unroll
    for (int t = 0; t < 4; ++t) {
      const int n = n0 + t * 16 + lo;
      const int h = n >> 6, d = n & 63;
#pragma unroll
      for (int r = 0; r < 8; ++r) {
        const int M = row0 + r + hi * 8;
        const int srow = M & (CS - 1);
        const int bpos = M >> 11;
        obf[((size_t)(bpos * CH + h) * CD + d) * CS + srow] = (__bf16)acc[t][r];
      }
    }
  } else {  // f32 row-major out
#pragma unroll
    for (int t = 0; t < 4; ++t) {
      const int n = n0 + t * 16 + lo;
#pragma unroll
      for (int r = 0; r < 8; ++r) {
        const int M = row0 + r + hi * 8;
        ofl[(size_t)M * CHID + n] = acc[t][r];
      }
    }
  }
}

// ---------------------------------------------------------------------------
// Causal flash attention: one wave owns a 16-row q tile; online softmax;
// Q[B][H][S][D], K[B][H][S][D], Vt[B][H][D][S] (all bf16) -> O bf16 [B][S][HID]
// Block 256 = 8 waves; grid (S/128, B*H)
// ---------------------------------------------------------------------------
__global__ void __launch_bounds__(256) flash_attn(
    const __bf16* __restrict__ Qb, const __bf16* __restrict__ Kb,
    const __bf16* __restrict__ Vtb, __bf16* __restrict__ Ob) {
  __shared__ __align__(32) __bf16 pbuf[8][16 * 32];  // per-wave P staging (C->A layout)

  const int lane = threadIdx.x & 31;
  const int wv   = threadIdx.x >> 5;
  const int lo   = lane & 15;
  const int hi   = lane >> 4;
  const int qbase = blockIdx.x * 128 + wv * 16;
  const int bh = blockIdx.y;
  const int bpos = bh >> 4;
  const int h    = bh & 15;

  const __bf16* Qh = Qb  + (size_t)bh * CS * CD;
  const __bf16* Kh = Kb  + (size_t)bh * CS * CD;
  const __bf16* Vh = Vtb + (size_t)bh * CD * CS;

  // Q tile in A layout, two K-dim chunks covering D=64
  v16bf qa[2];
  {
    const __bf16* qrow = Qh + (size_t)(qbase + lo) * CD + hi * 16;
    qa[0] = *(const v16bf*)(qrow);
    qa[1] = *(const v16bf*)(qrow + 32);
  }

  v8f o[4] = {};
  float mrow[8], lrow[8];
#pragma unroll
  for (int r = 0; r < 8; ++r) { mrow[r] = -3.0e38f; lrow[r] = 0.f; }

  const int kend = qbase + 16;  // causal: rows qbase..qbase+15 need k <= qbase+15
  for (int k0 = 0; k0 < kend; k0 += 32) {
    // ---- scores S = Q * K^T (two 16x16 N tiles, K-dim = D = 64) ----
    v8f s0 = {}, s1 = {};
    {
      int kr0 = k0 + lo;        if (kr0 >= CS) kr0 = CS - 1;
      int kr1 = k0 + 16 + lo;   if (kr1 >= CS) kr1 = CS - 1;
      const __bf16* kp0 = Kh + (size_t)kr0 * CD + hi * 16;
      const __bf16* kp1 = Kh + (size_t)kr1 * CD + hi * 16;
      s0 = wmma_bf16(qa[0], *(const v16bf*)(kp0),      s0);
      s0 = wmma_bf16(qa[1], *(const v16bf*)(kp0 + 32), s0);
      s1 = wmma_bf16(qa[0], *(const v16bf*)(kp1),      s1);
      s1 = wmma_bf16(qa[1], *(const v16bf*)(kp1 + 32), s1);
    }

    // ---- online softmax (row = r + hi*8; cols striped across 16-lane half) ----
#pragma unroll
    for (int r = 0; r < 8; ++r) {
      const int q   = qbase + r + hi * 8;
      const int kc0 = k0 + lo;
      const int kc1 = k0 + 16 + lo;
      float v0 = s0[r] * 0.125f + (kc0 > q ? -1.0e9f : 0.f);  // D^-0.5 = 0.125
      float v1 = s1[r] * 0.125f + (kc1 > q ? -1.0e9f : 0.f);
      float mx = fmaxf(v0, v1);
      mx = fmaxf(mx, __shfl_xor(mx, 1, 32));
      mx = fmaxf(mx, __shfl_xor(mx, 2, 32));
      mx = fmaxf(mx, __shfl_xor(mx, 4, 32));
      mx = fmaxf(mx, __shfl_xor(mx, 8, 32));
      const float mn = fmaxf(mrow[r], mx);
      const float alpha = __expf(mrow[r] - mn);
      v0 = __expf(v0 - mn);
      v1 = __expf(v1 - mn);
      float rs = v0 + v1;
      rs += __shfl_xor(rs, 1, 32);
      rs += __shfl_xor(rs, 2, 32);
      rs += __shfl_xor(rs, 4, 32);
      rs += __shfl_xor(rs, 8, 32);
      lrow[r] = lrow[r] * alpha + rs;
      mrow[r] = mn;
#pragma unroll
      for (int t = 0; t < 4; ++t) o[t][r] = o[t][r] * alpha;
      // stage P (C layout) into LDS row-major 16x32 for re-load in A layout
      pbuf[wv][(r + hi * 8) * 32 + lo]      = (__bf16)v0;
      pbuf[wv][(r + hi * 8) * 32 + 16 + lo] = (__bf16)v1;
    }

    asm volatile("s_wait_dscnt 0x0" ::: "memory");

    // ---- P in A layout (16x32) ----
    const __bf16* prow = &pbuf[wv][lo * 32 + hi * 16];
    v16bf pA = *(const v16bf*)(prow);

    // ---- O += P * V  (V^T row-major: per-lane contiguous K) ----
    int kv = k0 + hi * 16;
    if (kv > CS - 16) kv = CS - 16;  // clamp: P is exactly 0 for masked cols
#pragma unroll
    for (int t = 0; t < 4; ++t) {
      const __bf16* vp = Vh + (size_t)(t * 16 + lo) * CS + kv;
      o[t] = wmma_bf16(pA, *(const v16bf*)(vp), o[t]);
    }
  }

  // ---- normalize and write O as [B][S][HID] bf16 ----
#pragma unroll
  for (int t = 0; t < 4; ++t) {
#pragma unroll
    for (int r = 0; r < 8; ++r) {
      const int q = qbase + r + hi * 8;
      const float val = o[t][r] / lrow[r];
      Ob[((size_t)bpos * CS + q) * CHID + h * CD + t * 16 + lo] = (__bf16)val;
    }
  }
}

// ---------------------------------------------------------------------------
// Launch
// ---------------------------------------------------------------------------
extern "C" void kernel_launch(void* const* d_in, const int* in_sizes, int n_in,
                              void* d_out, int out_size, void* d_ws, size_t ws_size,
                              hipStream_t stream) {
  (void)in_sizes; (void)n_in; (void)out_size; (void)ws_size;
  const float* hs   = (const float*)d_in[0];
  const float* fcos = (const float*)d_in[1];
  const float* fsin = (const float*)d_in[2];
  // d_in[3] = attn_mask: causal triu(-1e9,k=1) — reproduced analytically in-kernel
  const float* Wq = (const float*)d_in[4];
  const float* Wk = (const float*)d_in[5];
  const float* Wv = (const float*)d_in[6];
  const float* Wo = (const float*)d_in[7];

  char* ws = (char*)d_ws;
  const size_t MB = 1u << 20;
  __bf16* hsb = (__bf16*)(ws);              // 8 MB  : hidden bf16 [B*S][HID]
  __bf16* Wqb = (__bf16*)(ws + 8  * MB);    // 2 MB
  __bf16* Wkb = (__bf16*)(ws + 10 * MB);    // 2 MB
  __bf16* Wvb = (__bf16*)(ws + 12 * MB);    // 2 MB
  __bf16* Wob = (__bf16*)(ws + 14 * MB);    // 2 MB
  __bf16* Qb  = (__bf16*)(ws + 16 * MB);    // 8 MB  : [B][H][S][D]
  __bf16* Kb  = (__bf16*)(ws + 24 * MB);    // 8 MB  : [B][H][S][D]
  __bf16* Vtb = (__bf16*)(ws + 32 * MB);    // 8 MB  : [B][H][D][S]
  __bf16* Ob  = (__bf16*)(ws + 40 * MB);    // 8 MB  : [B][S][HID]

  const int nHS = CB * CS * CHID;
  const int nW  = CHID * CHID;
  cvt_f32_bf16<<<2048, 256, 0, stream>>>(hs, hsb, nHS);
  cvt_f32_bf16<<<1024, 256, 0, stream>>>(Wq, Wqb, nW);
  cvt_f32_bf16<<<1024, 256, 0, stream>>>(Wk, Wkb, nW);
  cvt_f32_bf16<<<1024, 256, 0, stream>>>(Wv, Wvb, nW);
  cvt_f32_bf16<<<1024, 256, 0, stream>>>(Wo, Wob, nW);

  dim3 gp(CB * CS / 128, CHID / 64);  // (32, 16)
  proj_gemm<<<gp, 256, 0, stream>>>(hsb, Wqb, Qb,  nullptr, fcos, fsin, 0);
  proj_gemm<<<gp, 256, 0, stream>>>(hsb, Wkb, Kb,  nullptr, fcos, fsin, 0);
  proj_gemm<<<gp, 256, 0, stream>>>(hsb, Wvb, Vtb, nullptr, fcos, fsin, 1);

  flash_attn<<<dim3(CS / 128, CB * CH), 256, 0, stream>>>(Qb, Kb, Vtb, Ob);

  proj_gemm<<<gp, 256, 0, stream>>>(Ob, Wob, nullptr, (float*)d_out, fcos, fsin, 2);
}
